// KmeansAttention_64450279244053
// MI455X (gfx1250) — compile-verified, matrix operand order checked by reading, not run on hardware
//
#include <hip/hip_runtime.h>

// ---------------------------------------------------------------------------
// KmeansAttention (routing attention) for MI455X / gfx1250, wave32 + WMMA.
// All matmuls use v_wmma_f32_16x16x32_f16 (f16 in, f32 accumulate).
// Workspace layout (bytes):
//   [0, 67108864)           dists  f32 [BH][NC][T]
//   [67108864, 68157440)    indices i32 [BH][NC][WSZ]
//   [68157440, 69206016)    denom  f32 [BH][T]
//   [69206016, 69206020)    loss   f32 [1]
// ---------------------------------------------------------------------------

#define BB   4
#define HH   8
#define TT   8192
#define DD   64
#define NCL  64
#define WSZ  128
#define BH   (BB * HH)
#define NTOT (BB * HH * TT * DD)        // 16777216 output elements (+1 loss)
#define SCALE 0.125f                    // D^-0.5
#define NEG_BIG -50000.0f

typedef __attribute__((ext_vector_type(16))) _Float16 v16h;
typedef __attribute__((ext_vector_type(8)))  float    v8f;

__device__ __forceinline__ v8f wmma16(v16h a, v16h b, v8f c) {
  // D = A(16x32 f16) * B(32x16 f16) + C(16x16 f32)
  return __builtin_amdgcn_wmma_f32_16x16x32_f16(false, a, false, b,
                                                (short)0, c, false, false);
}

// Load a 16x32 f16 A-fragment (or the symmetric per-lane-column B-fragment)
// from a row-major f16 row pointer. Per ISA 7.12.2: lane half selects K+8,
// VGPRs 0..3 hold K 0..7 (pairs), VGPRs 4..7 hold K 16..23 (pairs).
__device__ __forceinline__ v16h load_frag(const _Float16* row, int kc, int lane) {
  const _Float16* p = row + kc + ((lane >> 4) & 1) * 8;
  v16h f;
#pragma unroll
  for (int v = 0; v < 8; ++v) {
    int kk = ((v & 4) ? 16 : 0) + (v & 3) * 2;
    f[2 * v]     = p[kk];
    f[2 * v + 1] = p[kk + 1];
  }
  return f;
}

// ---------------------------------------------------------------------------
__global__ void k_init(float* out, int out_n, float* denom, float* loss) {
  size_t i = (size_t)blockIdx.x * blockDim.x + threadIdx.x;
  size_t stride = (size_t)gridDim.x * blockDim.x;
  for (size_t j = i; j < (size_t)out_n; j += stride) out[j] = 0.f;
  for (size_t j = i; j < (size_t)BH * TT; j += stride) denom[j] = 0.f;
  if (i == 0) loss[0] = 0.f;
}

// ---------------------------------------------------------------------------
// dists[bh][c][t] = dot(l2norm(qk[bh][t]), means[h][c]); also argmax bucket
// per token -> commitment-loss partial. One block per (bh, 128-token tile).
__global__ __launch_bounds__(256) void k_dists(const float* __restrict__ qk,
                                               const float* __restrict__ means,
                                               float* __restrict__ dist_g,
                                               float* __restrict__ loss) {
  __shared__ __align__(16) unsigned char smem[107520];
  float*    kr32 = (float*)smem;                 // 128x64 f32  (32768)
  _Float16* kr16 = (_Float16*)(smem + 32768);    // 128x64 f16  (16384)
  _Float16* mn16 = (_Float16*)(smem + 49152);    // 64x64  f16  ( 8192)
  float*    mn32 = (float*)(smem + 57344);       // 64x64  f32  (16384)
  float*    dloc = (float*)(smem + 73728);       // 128x64 f32  (32768)
  float*    red  = (float*)(smem + 106496);      // 256    f32  ( 1024)

  int tid = threadIdx.x;
  int blk = blockIdx.x;          // bh * (T/128) + tile
  int bh  = blk >> 6;
  int tb  = blk & 63;
  int h   = bh & (HH - 1);
  int t0  = tb * WSZ;

  // stage means (f16 for WMMA, f32 for loss)
  for (int e = tid; e < NCL * DD; e += 256) {
    float mv = means[(size_t)h * NCL * DD + e];
    mn32[e] = mv;
    mn16[e] = (_Float16)mv;
  }
  // stage normalized rows: 2 threads per row (32 dims each)
  {
    int m  = tid >> 1;
    int d0 = (tid & 1) * 32;
    const float* qrow = qk + ((size_t)bh * TT + t0 + m) * DD;
    float tmp[32];
    float ss = 0.f;
#pragma unroll
    for (int d = 0; d < 32; ++d) { float x = qrow[d0 + d]; tmp[d] = x; ss += x * x; }
    ss += __shfl_xor(ss, 1);
    float inv = 1.0f / fmaxf(sqrtf(ss), 1e-12f);
#pragma unroll
    for (int d = 0; d < 32; ++d) {
      float xr = tmp[d] * inv;
      kr32[m * DD + d0 + d] = xr;
      kr16[m * DD + d0 + d] = (_Float16)xr;
    }
  }
  __syncthreads();

  // WMMA: 128x64 = (128x64) @ (64x64)^T ; wave w owns rows 16w..16w+15
  int lane = tid & 31, wv = tid >> 5, half = (lane >> 4) & 1, nl = lane & 15;
  int mb = wv * 16;
  const _Float16* arow = kr16 + (mb + nl) * DD;
  v16h a0 = load_frag(arow, 0, lane);
  v16h a1 = load_frag(arow, 32, lane);
#pragma unroll
  for (int nb = 0; nb < 4; ++nb) {
    const _Float16* brow = mn16 + (nb * 16 + nl) * DD;
    v16h b0 = load_frag(brow, 0, lane);
    v16h b1 = load_frag(brow, 32, lane);
    v8f acc = {};
    acc = wmma16(a0, b0, acc);
    acc = wmma16(a1, b1, acc);
#pragma unroll
    for (int r = 0; r < 8; ++r) {
      int mm = mb + r + half * 8;
      int nn = nb * 16 + nl;
      float dv = acc[r];
      dloc[mm * NCL + nn] = dv;
      dist_g[((size_t)bh * NCL + nn) * TT + t0 + mm] = dv;
    }
  }
  __syncthreads();

  // argmax bucket + commitment loss partial
  float lv = 0.f;
  if (tid < WSZ) {
    int row = tid;
    float best = dloc[row * NCL];
    int bc = 0;
    for (int c = 1; c < NCL; ++c) {
      float x = dloc[row * NCL + c];
      if (x > best) { best = x; bc = c; }
    }
    float s2 = 0.f;
    for (int d = 0; d < DD; ++d) {
      float df = kr32[row * DD + d] - mn32[bc * DD + d];
      s2 += df * df;
    }
    lv = s2;
  }
  red[tid] = lv;
  for (int s = 128; s > 0; s >>= 1) {
    __syncthreads();
    if (tid < s) red[tid] += red[tid + s];
  }
  __syncthreads();
  if (tid == 0) atomicAdd(loss, red[0]);
}

// ---------------------------------------------------------------------------
// Top-128 per (bh, cluster) with jax tie rule (equal value -> lower index),
// then ascending bitonic sort of the selected indices.
__global__ __launch_bounds__(256) void k_topk(const float* __restrict__ dist_g,
                                              int* __restrict__ idx_g) {
  __shared__ float vals[TT];
  __shared__ float rv[256];
  __shared__ int   ri[256];
  __shared__ int   sel[WSZ];
  int tid = threadIdx.x;
  int blk = blockIdx.x;                 // bh * NCL + c
  const float* src = dist_g + (size_t)blk * TT;
  for (int i = tid; i < TT; i += 256) vals[i] = src[i];
  __syncthreads();

  for (int it = 0; it < WSZ; ++it) {
    float best = -3.4e38f; int bi = TT;
    for (int i = tid; i < TT; i += 256) {
      float x = vals[i];
      if (x > best) { best = x; bi = i; }   // first hit keeps lowest index
    }
    rv[tid] = best; ri[tid] = bi;
    for (int s = 128; s > 0; s >>= 1) {
      __syncthreads();
      if (tid < s) {
        float xv = rv[tid + s]; int xi = ri[tid + s];
        if (xv > rv[tid] || (xv == rv[tid] && xi < ri[tid])) { rv[tid] = xv; ri[tid] = xi; }
      }
    }
    __syncthreads();
    if (tid == 0) { sel[it] = ri[0]; vals[ri[0]] = -3.4e38f; }
    __syncthreads();
  }

  // ascending bitonic sort, 128 elements
  for (int k2 = 2; k2 <= WSZ; k2 <<= 1)
    for (int j = k2 >> 1; j > 0; j >>= 1) {
      __syncthreads();
      if (tid < WSZ) {
        int ixj = tid ^ j;
        if (ixj > tid) {
          int a = sel[tid], b2 = sel[ixj];
          bool sw = ((tid & k2) == 0) ? (a > b2) : (a < b2);
          if (sw) { sel[tid] = b2; sel[ixj] = a; }
        }
      }
    }
  __syncthreads();
  if (tid < WSZ) idx_g[(size_t)blk * WSZ + tid] = sel[tid];
}

// ---------------------------------------------------------------------------
// Per-cluster windowed attention. One block (8 wave32) per (bh, cluster).
__global__ __launch_bounds__(256) void k_attn(const float* __restrict__ qk,
                                              const float* __restrict__ vin,
                                              const float* __restrict__ relw,
                                              const int* __restrict__ idx_g,
                                              float* __restrict__ out,
                                              float* __restrict__ denom) {
  __shared__ __align__(16) unsigned char smem[131584];
  _Float16* q16  = (_Float16*)(smem);             // 128x64  f16 (16384)
  _Float16* k16  = (_Float16*)(smem + 16384);     // 128x64  f16 (16384)
  _Float16* vT   = (_Float16*)(smem + 32768);     // 64x128  f16 (16384) V^T
  _Float16* r16  = (_Float16*)(smem + 49152);     // 128x64  f16 (16384)
  _Float16* rels = (_Float16*)(smem + 65536);     // 128x128 f16 (32768)
  _Float16* p16  = (_Float16*)(smem + 98304);     // 128x128 f16 (32768)
  float*    qstg = (float*)(smem + 98304);        // alias of p16 (staging)
  int*      idx  = (int*)(smem + 131072);         // 128 i32

  int tid = threadIdx.x;
  int blk = blockIdx.x;                 // bh * NCL + c
  int bh  = blk >> 6;
  int h   = bh & (HH - 1);

  if (tid < WSZ) idx[tid] = idx_g[(size_t)blk * WSZ + tid];
  __syncthreads();

  // ---- gather + normalize + transpose-stage (2 threads / row) ----
  {
    int m  = tid >> 1;
    int d0 = (tid & 1) * 32;
    int tok = idx[m];
    const float* qrow = qk  + ((size_t)bh * TT + tok) * DD;
    const float* vrow = vin + ((size_t)bh * TT + tok) * DD;
    float ss = 0.f;
#pragma unroll
    for (int d = 0; d < 32; ++d) {
      float x = qrow[d0 + d];
      qstg[m * DD + d0 + d] = x;
      q16[m * DD + d0 + d]  = (_Float16)x;
      ss += x * x;
    }
    ss += __shfl_xor(ss, 1);
    float inv = 1.0f / fmaxf(sqrtf(ss), 1e-12f);
#pragma unroll
    for (int d = 0; d < 32; ++d) {
      k16[m * DD + d0 + d]  = (_Float16)(qstg[m * DD + d0 + d] * inv);
      vT[(d0 + d) * WSZ + m] = (_Float16)vrow[d0 + d];
      r16[m * DD + d0 + d]  = (_Float16)relw[((size_t)m * HH + h) * DD + d0 + d];
    }
  }
  __syncthreads();

  int lane = tid & 31, wv = tid >> 5, half = (lane >> 4) & 1, nl = lane & 15;
  int mb = wv * 16;
  const _Float16* arow = q16 + (mb + nl) * DD;
  v16h aq0 = load_frag(arow, 0, lane);
  v16h aq1 = load_frag(arow, 32, lane);

  // ---- Rel = Q @ RelW^T -> LDS (f16) ----
#pragma unroll
  for (int nb = 0; nb < 8; ++nb) {
    const _Float16* brow = r16 + (nb * 16 + nl) * DD;
    v16h b0 = load_frag(brow, 0, lane);
    v16h b1 = load_frag(brow, 32, lane);
    v8f acc = {};
    acc = wmma16(aq0, b0, acc);
    acc = wmma16(aq1, b1, acc);
#pragma unroll
    for (int r = 0; r < 8; ++r)
      rels[(mb + r + half * 8) * WSZ + nb * 16 + nl] = (_Float16)acc[r];
  }

  // ---- S = Q @ K^T (kept in registers) ----
  v8f sacc[8];
#pragma unroll
  for (int nb = 0; nb < 8; ++nb) {
    const _Float16* brow = k16 + (nb * 16 + nl) * DD;
    v16h b0 = load_frag(brow, 0, lane);
    v16h b1 = load_frag(brow, 32, lane);
    v8f acc = {};
    acc = wmma16(aq0, b0, acc);
    acc = wmma16(aq1, b1, acc);
    sacc[nb] = acc;
  }
  __syncthreads();   // rels visible

  // ---- scale + shifted rel + diag mask + row softmax ----
  float rmax[8], rsum[8];
#pragma unroll
  for (int r = 0; r < 8; ++r) rmax[r] = -3.4e38f;
#pragma unroll
  for (int nb = 0; nb < 8; ++nb) {
#pragma unroll
    for (int r = 0; r < 8; ++r) {
      int mm = mb + r + half * 8;
      int nn = nb * 16 + nl;
      float x = sacc[nb][r] * SCALE;
      if (nn <= mm) x += SCALE * (float)rels[mm * WSZ + (nn - mm + (WSZ - 1))];
      if (nn == mm) x = NEG_BIG;
      sacc[nb][r] = x;
      rmax[r] = fmaxf(rmax[r], x);
    }
  }
#pragma unroll
  for (int r = 0; r < 8; ++r) {
#pragma unroll
    for (int msk = 1; msk < 16; msk <<= 1)
      rmax[r] = fmaxf(rmax[r], __shfl_xor(rmax[r], msk));
    rsum[r] = 0.f;
  }
#pragma unroll
  for (int nb = 0; nb < 8; ++nb)
#pragma unroll
    for (int r = 0; r < 8; ++r) {
      float e = __expf(sacc[nb][r] - rmax[r]);
      sacc[nb][r] = e;
      rsum[r] += e;
    }
#pragma unroll
  for (int r = 0; r < 8; ++r) {
#pragma unroll
    for (int msk = 1; msk < 16; msk <<= 1)
      rsum[r] += __shfl_xor(rsum[r], msk);
    rsum[r] = 1.0f / rsum[r];
  }
#pragma unroll
  for (int nb = 0; nb < 8; ++nb)
#pragma unroll
    for (int r = 0; r < 8; ++r)
      p16[(mb + r + half * 8) * WSZ + nb * 16 + nl] = (_Float16)(sacc[nb][r] * rsum[r]);
  __syncthreads();

  // ---- O = P @ V, scatter-add into numer/denom ----
  v16h ap[4];
#pragma unroll
  for (int kc = 0; kc < 4; ++kc)
    ap[kc] = load_frag(p16 + (mb + nl) * WSZ, kc * 32, lane);
#pragma unroll
  for (int nb = 0; nb < 4; ++nb) {
    v8f o = {};
#pragma unroll
    for (int kc = 0; kc < 4; ++kc) {
      v16h bf = load_frag(vT + (nb * 16 + nl) * WSZ, kc * 32, lane);
      o = wmma16(ap[kc], bf, o);
    }
#pragma unroll
    for (int r = 0; r < 8; ++r) {
      int mm = mb + r + half * 8;
      int dd = nb * 16 + nl;
      int tk = idx[mm];
      atomicAdd(&out[((size_t)bh * TT + tk) * DD + dd], o[r]);
    }
  }
  if (tid < WSZ) atomicAdd(&denom[(size_t)bh * TT + idx[tid]], 1.0f);
}

// ---------------------------------------------------------------------------
__global__ void k_final(float* out, const float* __restrict__ denom,
                        const float* __restrict__ loss) {
  size_t i = (size_t)blockIdx.x * blockDim.x + threadIdx.x;
  size_t stride = (size_t)gridDim.x * blockDim.x;
  for (size_t j = i; j < (size_t)NTOT; j += stride)
    out[j] = out[j] / (denom[j >> 6] + 1e-5f);
  if (i == 0) out[NTOT] = loss[0] * (0.0001f / (float)NTOT);
}

// ---------------------------------------------------------------------------
extern "C" void kernel_launch(void* const* d_in, const int* in_sizes, int n_in,
                              void* d_out, int out_size, void* d_ws, size_t ws_size,
                              hipStream_t stream) {
  const float* qk    = (const float*)d_in[0];
  const float* v     = (const float*)d_in[1];
  const float* means = (const float*)d_in[2];
  const float* relw  = (const float*)d_in[3];
  float* out = (float*)d_out;
  char*  ws  = (char*)d_ws;

  float* dists = (float*)ws;                    // 67108864 B
  int*   idxs  = (int*)(ws + 67108864);         //  1048576 B
  float* denom = (float*)(ws + 68157440);       //  1048576 B
  float* loss  = (float*)(ws + 69206016);       //        4 B

  (void)in_sizes; (void)n_in; (void)ws_size;

  k_init <<<2048, 256, 0, stream>>>(out, out_size, denom, loss);
  k_dists<<<BH * (TT / WSZ), 256, 0, stream>>>(qk, means, dists, loss);
  k_topk <<<BH * NCL, 256, 0, stream>>>(dists, idxs);
  k_attn <<<BH * NCL, 256, 0, stream>>>(qk, v, relw, idxs, out, denom);
  k_final<<<4096, 256, 0, stream>>>(out, denom, loss);
}
